// compute_dnl_53214644798217
// MI455X (gfx1250) — matrix operand order repaired and checked by reading, not compiled, not run on hardware
//
#include <hip/hip_runtime.h>

typedef __attribute__((ext_vector_type(16))) __bf16       v16bf;
typedef __attribute__((ext_vector_type(8)))  float        v8f;
typedef __attribute__((ext_vector_type(4)))  unsigned int v4u;

#define NN    64
#define CC    192
#define VV    25
#define TTOT  300
#define DDIM  192
#define TB    8
#define NTB   38      /* ceil(300/8) */
#define CP    200     /* padded row length (bf16) of XB rows  */
#define DP    200     /* padded row length (bf16) of G rows   */
#define JROWS 208     /* XB rows (200 used + 8 zero pad)      */
#define GROWS 256     /* TB*32 rows for G1/G2                 */

union Frag { v4u q[2]; v16bf v; };

__device__ __forceinline__ unsigned short f2bf(float f) {
  unsigned int u = __float_as_uint(f);
  u += 0x7FFFu + ((u >> 16) & 1u);          // round-to-nearest-even
  return (unsigned short)(u >> 16);
}
__device__ __forceinline__ float bf2f(unsigned short h) {
  return __uint_as_float(((unsigned int)h) << 16);
}
__device__ __forceinline__ v8f wmma_bf16(v16bf a, v16bf b, v8f c) {
  return __builtin_amdgcn_wmma_f32_16x16x32_bf16(false, a, false, b, (short)0, c, false, false);
}
__device__ __forceinline__ v8f vzero8() {
  v8f z; for (int q = 0; q < 8; ++q) z[q] = 0.f; return z;
}

__global__ void prep_w_kernel(const float* __restrict__ w1, const float* __restrict__ w2,
                              unsigned short* __restrict__ w1b, unsigned short* __restrict__ w2b) {
  int i = blockIdx.x * blockDim.x + threadIdx.x;
  if (i < DDIM * CC) { w1b[i] = f2bf(w1[i]); w2b[i] = f2bf(w2[i]); }
}

__global__ void __launch_bounds__(256, 1)
fused_attn_kernel(const float* __restrict__ x1,
                  const unsigned short* __restrict__ w1b,
                  const unsigned short* __restrict__ w2b,
                  const float* __restrict__ wmask,
                  float* __restrict__ out) {
  extern __shared__ char smem_raw[];
  unsigned short* XB = (unsigned short*)smem_raw;   // [JROWS][CP]  x block, bf16
  unsigned short* G1 = XB + JROWS * CP;             // [GROWS][DP]  proj1, bf16
  unsigned short* G2 = G1 + GROWS * DP;             // [GROWS][DP]  proj2, bf16
  float* MASKL = (float*)(G2 + GROWS * DP);         // [GROWS]      mask per (tt,v)

  const int tid  = threadIdx.x;
  const int lane = tid & 31;
  const int wid  = tid >> 5;
  const int ln   = lane & 15;
  const bool hi  = lane >= 16;
  const int n    = blockIdx.x / NTB;
  const int t0   = (blockIdx.x % NTB) * TB;

  // ---- zero pad regions (XB rows 200..207; G rows with v in 25..31) ----
  for (int i = tid; i < 8 * CP; i += 256) XB[200 * CP + i] = 0;
  for (int i = tid; i < 56 * DP; i += 256) {
    int rr = i / DP, cc = i % DP;
    int row = (rr / 7) * 32 + 25 + (rr % 7);
    G1[row * DP + cc] = 0;
    G2[row * DP + cc] = 0;
  }
  // ---- stage x1 block into LDS as bf16: XB[(tt*25+v)][c] ----
  for (int i = tid; i < CC * VV * TB; i += 256) {
    int tt = i % TB;
    int v  = (i / TB) % VV;
    int c  = i / (TB * VV);
    int t  = t0 + tt;
    float x = (t < TTOT) ? x1[(((size_t)n * CC + c) * VV + v) * TTOT + t] : 0.f;
    XB[(tt * VV + v) * CP + c] = f2bf(x);
  }
  __syncthreads();

  // ---- mask[n,t,v] = sum_c wmask[c] * x[n,c,v,t] ----
  if (tid < VV * TB) {
    int ttm = tid / VV, vm = tid % VV;
    const unsigned short* xr = &XB[tid * CP];
    float s = 0.f;
    for (int c = 0; c < CC; ++c) s += bf2f(xr[c]) * wmask[c];
    MASKL[ttm * 32 + vm] = s;
  }

  // ---- stage 1: G = W * X  (M=192, N=200->13 tiles, K=192) for both W1,W2 ----
  // chunk = (n-tile, m-half): 13 n-tiles x 2 halves of 6 m-tiles
  for (int ch = wid; ch < 26; ch += 8) {
    int nt = ch % 13;
    int mh = ch / 13;
    int jbase = nt * 16;
    v8f acc1[6], acc2[6];
    for (int mt = 0; mt < 6; ++mt) { acc1[mt] = vzero8(); acc2[mt] = vzero8(); }
    for (int ks = 0; ks < 6; ++ks) {
      int kb = ks * 32;
      // B frag (16-bit, 32x16): lanes 0-15 -> N=lane, K{kb..+7, kb+16..+23}; lanes 16-31 +8
      const unsigned short* bp = &XB[(jbase + ln) * CP + kb + (hi ? 8 : 0)];
      Frag bfr; bfr.q[0] = *(const v4u*)bp; bfr.q[1] = *(const v4u*)(bp + 16);
      for (int mt = 0; mt < 6; ++mt) {
        int m = (mh * 6 + mt) * 16 + ln;
        // A frag (16-bit, 16x32): lanes0-15 K{kb..+7, kb+16..+23}; lanes16-31 +8
        const unsigned short* a1p = &w1b[m * CC + kb + (hi ? 8 : 0)];
        Frag a1; a1.q[0] = *(const v4u*)a1p; a1.q[1] = *(const v4u*)(a1p + 16);
        acc1[mt] = wmma_bf16(a1.v, bfr.v, acc1[mt]);
        const unsigned short* a2p = &w2b[m * CC + kb + (hi ? 8 : 0)];
        Frag a2; a2.q[0] = *(const v4u*)a2p; a2.q[1] = *(const v4u*)(a2p + 16);
        acc2[mt] = wmma_bf16(a2.v, bfr.v, acc2[mt]);
      }
    }
    // write C tiles to LDS as G[row=(tt*32+v)][d], bf16, 16B per lane per tile
    int j = jbase + ln;
    if (j < VV * TB) {
      int grow = (j / VV) * 32 + (j % VV);
      for (int mt = 0; mt < 6; ++mt) {
        int mb = (mh * 6 + mt) * 16 + (hi ? 8 : 0);   // lane's 8 consecutive d's
        v4u p1, p2;
        for (int q = 0; q < 4; ++q) {
          p1[q] = (unsigned int)f2bf(acc1[mt][2 * q]) |
                  ((unsigned int)f2bf(acc1[mt][2 * q + 1]) << 16);
          p2[q] = (unsigned int)f2bf(acc2[mt][2 * q]) |
                  ((unsigned int)f2bf(acc2[mt][2 * q + 1]) << 16);
        }
        *(v4u*)&G1[grow * DP + mb] = p1;
        *(v4u*)&G2[grow * DP + mb] = p2;
      }
    }
  }
  __syncthreads();

  // ---- centering: subtract mean over v for each (tt, d) ----
  for (int p = tid; p < TB * DDIM; p += 256) {
    int d = p % DDIM, tt = p / DDIM;
    unsigned short* g1 = &G1[(tt * 32) * DP + d];
    unsigned short* g2 = &G2[(tt * 32) * DP + d];
    float s1 = 0.f, s2 = 0.f;
    for (int v = 0; v < VV; ++v) { s1 += bf2f(g1[v * DP]); s2 += bf2f(g2[v * DP]); }
    float m1 = s1 * (1.f / VV), m2 = s2 * (1.f / VV);
    for (int v = 0; v < VV; ++v) {
      g1[v * DP] = f2bf(bf2f(g1[v * DP]) - m1);
      g2[v * DP] = f2bf(bf2f(g2[v * DP]) - m2);
    }
  }
  __syncthreads();

  // ---- stage 2: S = G1c^T * G2c / sqrt(D) + mask, softmax over w. One t per wave. ----
  {
    const int tt = wid;
    const int t  = t0 + tt;
    const float scale = 0.0721687836487032f;   // 1/sqrt(192)
    const float mk0 = MASKL[tt * 32 + ln];
    const float mk1 = (ln < 9) ? MASKL[tt * 32 + 16 + ln] : 0.f;
    for (int mt = 0; mt < 2; ++mt) {
      v8f s0 = vzero8(), s1 = vzero8();
      for (int ks = 0; ks < 6; ++ks) {
        int kb = ks * 32;
        // A frag: rows = v (G1 rows), K = d
        const unsigned short* ap = &G1[(tt * 32 + mt * 16 + ln) * DP + kb + (hi ? 8 : 0)];
        Frag af; af.q[0] = *(const v4u*)ap; af.q[1] = *(const v4u*)(ap + 16);
        // B frags: B[k=d][n=w] = G2c[w][d] -> contiguous per-lane row reads
        const unsigned short* bp0 = &G2[(tt * 32 + ln) * DP + kb + (hi ? 8 : 0)];
        Frag b0; b0.q[0] = *(const v4u*)bp0; b0.q[1] = *(const v4u*)(bp0 + 16);
        s0 = wmma_bf16(af.v, b0.v, s0);
        const unsigned short* bp1 = &G2[(tt * 32 + 16 + ln) * DP + kb + (hi ? 8 : 0)];
        Frag b1; b1.q[0] = *(const v4u*)bp1; b1.q[1] = *(const v4u*)(bp1 + 16);
        s1 = wmma_bf16(af.v, b1.v, s1);
      }
      // softmax epilogue: row m lives in one 16-lane half across s0 (w=0..15) and s1 (w=16..24)
      for (int r = 0; r < 8; ++r) {
        float v0 = s0[r] * scale + mk0;
        float v1 = (ln < 9) ? (s1[r] * scale + mk1) : -3.0e38f;
        float mx = fmaxf(v0, v1);
        mx = fmaxf(mx, __shfl_xor(mx, 1));
        mx = fmaxf(mx, __shfl_xor(mx, 2));
        mx = fmaxf(mx, __shfl_xor(mx, 4));
        mx = fmaxf(mx, __shfl_xor(mx, 8));
        float e0 = __expf(v0 - mx);
        float e1 = (ln < 9) ? __expf(v1 - mx) : 0.f;
        float sm = e0 + e1;
        sm += __shfl_xor(sm, 1);
        sm += __shfl_xor(sm, 2);
        sm += __shfl_xor(sm, 4);
        sm += __shfl_xor(sm, 8);
        float inv = 1.f / sm;
        int vrow = mt * 16 + (hi ? 8 : 0) + r;
        if (vrow < VV && t < TTOT) {
          float* orow = out + ((((size_t)n * TTOT + t) * VV + vrow) * VV);
          orow[ln] = e0 * inv;
          if (ln < 9) orow[16 + ln] = e1 * inv;
        }
      }
    }
  }
}

extern "C" void kernel_launch(void* const* d_in, const int* in_sizes, int n_in,
                              void* d_out, int out_size, void* d_ws, size_t ws_size,
                              hipStream_t stream) {
  (void)in_sizes; (void)n_in; (void)out_size; (void)ws_size;
  const float* x1    = (const float*)d_in[0];
  const float* w1    = (const float*)d_in[1];
  const float* w2    = (const float*)d_in[2];
  const float* wmask = (const float*)d_in[3];

  unsigned short* w1b = (unsigned short*)d_ws;
  unsigned short* w2b = w1b + DDIM * CC;

  prep_w_kernel<<<(DDIM * CC + 255) / 256, 256, 0, stream>>>(w1, w2, w1b, w2b);

  size_t lds_bytes = (size_t)(JROWS * CP + 2 * GROWS * DP) * sizeof(unsigned short)
                   + GROWS * sizeof(float);    // 289,024 B < 320 KB
  fused_attn_kernel<<<NN * NTB, 256, lds_bytes, stream>>>(x1, w1b, w2b, wmask, (float*)d_out);
}
